// Net_11854109737607
// MI455X (gfx1250) — matrix operand order repaired and checked by reading, not compiled, not run on hardware
//
#include <hip/hip_runtime.h>
#include <stdint.h>

// ---------------- CDNA5 WMMA types ----------------
typedef __attribute__((ext_vector_type(16))) __bf16     v16bf;
typedef __attribute__((ext_vector_type(8)))  float      v8f;
typedef __attribute__((ext_vector_type(4)))  unsigned int u32x4;

__device__ __forceinline__ unsigned short f2bf(float f) {
  unsigned int u = __float_as_uint(f);
  return (unsigned short)((u + 0x7fffu + ((u >> 16) & 1u)) >> 16); // RNE
}

// ---------------- conversion kernels ----------------
__global__ __launch_bounds__(256) void k_convert_x(const float* __restrict__ x,
    unsigned short* __restrict__ xb, int N, int K, int Kp) {
  size_t i = (size_t)blockIdx.x * blockDim.x + threadIdx.x;
  if (i >= (size_t)N * Kp) return;
  int k = (int)(i % Kp);
  size_t n = i / Kp;
  float v = (k < K) ? x[n * (size_t)K + k] : 0.0f;
  xb[i] = f2bf(v);
}

// Wt[n*Kp + k] = bf16(W[k*Nw + n]); zero-pad both dims.
__global__ __launch_bounds__(256) void k_convert_w(const float* __restrict__ W,
    unsigned short* __restrict__ Wt, int K, int Nw, int Kp, int Np) {
  size_t i = (size_t)blockIdx.x * blockDim.x + threadIdx.x;
  if (i >= (size_t)Np * Kp) return;
  int k = (int)(i % Kp);
  int n = (int)(i / Kp);
  float v = (k < K && n < Nw) ? W[(size_t)k * Nw + n] : 0.0f;
  Wt[i] = f2bf(v);
}

// ---------------- WMMA GEMM: C[M,N] = A[M,K](bf16) * Bt[N,K](bf16), f32 accum ----------------
// 2x2 register-tiled: one wave computes a 32x32 C block (4 wmma per K-step from
// 2 A + 2 B fragments -> half the VMEM per wmma vs 1x1 tiling).
// Fragment layouts per CDNA5 ISA 7.12.2:
//  A 16x32: lane<16 row=lane, elems 0..7 -> K=kk+half*8.., elems 8..15 -> K=kk+16+half*8..
//  B 32x16 (from N-major transposed weights): lane col=lane&15, elems 0..7 -> K=kk+half*8.., elems 8..15 -> K=kk+16+half*8..
//  C 16x16: vgpr r -> M = r + half*8, col = lane&15.
__global__ __launch_bounds__(256) void k_gemm_wmma(
    const unsigned short* __restrict__ A, int lda,
    const unsigned short* __restrict__ Bt, int ldb,
    float* __restrict__ C, int ldc,
    int Mt2, int Nt2, int K) {
  int wave = blockIdx.x * (blockDim.x >> 5) + ((int)threadIdx.x >> 5);
  if (wave >= Mt2 * Nt2) return;
  int lane = threadIdx.x & 31;
  int half = lane >> 4;
  int l15  = lane & 15;
  // column-major wave->tile map: the 8 waves of a block share tn2 => B frags hit WGP$
  int tn2 = wave / Mt2;
  int tm2 = wave - tn2 * Mt2;

  const unsigned short* arow0 = A  + (size_t)(tm2 * 32 + l15) * lda + half * 8;
  const unsigned short* arow1 = arow0 + (size_t)16 * lda;
  const unsigned short* bcol0 = Bt + (size_t)(tn2 * 32 + l15) * ldb + half * 8;
  const unsigned short* bcol1 = bcol0 + (size_t)16 * ldb;

  v8f acc00 = {}, acc01 = {}, acc10 = {}, acc11 = {};
  for (int kk = 0; kk < K; kk += 32) {
    union { v16bf v; u32x4 q[2]; } a0, a1, b0, b1;
    a0.q[0] = *(const u32x4*)(arow0 + kk);
    a0.q[1] = *(const u32x4*)(arow0 + kk + 16);
    a1.q[0] = *(const u32x4*)(arow1 + kk);
    a1.q[1] = *(const u32x4*)(arow1 + kk + 16);
    b0.q[0] = *(const u32x4*)(bcol0 + kk);
    b0.q[1] = *(const u32x4*)(bcol0 + kk + 16);
    b1.q[0] = *(const u32x4*)(bcol1 + kk);
    b1.q[1] = *(const u32x4*)(bcol1 + kk + 16);
    if (kk + 128 < K) {
      __builtin_prefetch(arow0 + kk + 128, 0, 3);
      __builtin_prefetch(arow1 + kk + 128, 0, 3);
    }
    acc00 = __builtin_amdgcn_wmma_f32_16x16x32_bf16(false, a0.v, false, b0.v,
                                                    (short)0, acc00, false, false);
    acc01 = __builtin_amdgcn_wmma_f32_16x16x32_bf16(false, a0.v, false, b1.v,
                                                    (short)0, acc01, false, false);
    acc10 = __builtin_amdgcn_wmma_f32_16x16x32_bf16(false, a1.v, false, b0.v,
                                                    (short)0, acc10, false, false);
    acc11 = __builtin_amdgcn_wmma_f32_16x16x32_bf16(false, a1.v, false, b1.v,
                                                    (short)0, acc11, false, false);
  }
  float* c00 = C + (size_t)(tm2 * 32 + half * 8) * ldc + tn2 * 32 + l15;
#pragma unroll
  for (int r = 0; r < 8; ++r) {
    c00[(size_t)r * ldc]              = acc00[r];
    c00[(size_t)r * ldc + 16]         = acc01[r];
    c00[(size_t)(r + 16) * ldc]       = acc10[r];
    c00[(size_t)(r + 16) * ldc + 16]  = acc11[r];
  }
}

// ---------------- attention coefficients: es/ed[n,h] = sum_c h[n,h,c]*a[h,c] ----------------
__global__ __launch_bounds__(256) void k_attn_coef(
    const float* __restrict__ h, int ldh,
    const float* __restrict__ a_src, const float* __restrict__ a_dst,
    float* __restrict__ es, float* __restrict__ ed, int N, int H, int C) {
  int warp = blockIdx.x * (blockDim.x >> 5) + ((int)threadIdx.x >> 5);
  int lane = threadIdx.x & 31;
  int n = warp / H;
  int hh = warp - n * H;
  if (n >= N) return;
  const float* hp = h + (size_t)n * ldh + hh * C;
  const float* ap = a_src + hh * C;
  const float* dp = a_dst + hh * C;
  float s = 0.f, d = 0.f;
  for (int c = lane; c < C; c += 32) { float v = hp[c]; s += v * ap[c]; d += v * dp[c]; }
  for (int off = 16; off > 0; off >>= 1) { s += __shfl_xor(s, off); d += __shfl_xor(d, off); }
  if (lane == 0) { es[(size_t)n * 8 + hh] = s; ed[(size_t)n * 8 + hh] = d; }
}

__global__ __launch_bounds__(256) void k_fill(float* __restrict__ p, float v, size_t n) {
  size_t i = (size_t)blockIdx.x * blockDim.x + threadIdx.x;
  if (i < n) p[i] = v;
}

// ---------------- edge passes ----------------
__device__ __forceinline__ void edge_pair(const int* __restrict__ src,
    const int* __restrict__ dst, int E, int e, int& s, int& d) {
  if (e < E) { s = src[e]; d = dst[e]; } else { s = d = e - E; } // appended self-loops
}

__global__ __launch_bounds__(256) void k_edge_max(
    const int* __restrict__ src, const int* __restrict__ dst, int E, int Etot,
    const float* __restrict__ es, const float* __restrict__ ed,
    float* __restrict__ m, int H) {
  int e = blockIdx.x * blockDim.x + threadIdx.x;
  if (e >= Etot) return;
  int s, d; edge_pair(src, dst, E, e, s, d);
  for (int hh = 0; hh < H; ++hh) {
    float v = es[(size_t)s * 8 + hh] + ed[(size_t)d * 8 + hh];
    v = v > 0.f ? v : 0.2f * v;                 // leaky_relu(.,0.2)
    float* a = m + (size_t)d * 8 + hh;          // order-preserving f32 atomic max
    if (v >= 0.f) atomicMax((int*)a, __float_as_int(v));
    else          atomicMin((unsigned int*)a, __float_as_uint(v));
  }
}

__global__ __launch_bounds__(256) void k_edge_sum(
    const int* __restrict__ src, const int* __restrict__ dst, int E, int Etot,
    const float* __restrict__ es, const float* __restrict__ ed,
    const float* __restrict__ m, float* __restrict__ denom, int H) {
  int e = blockIdx.x * blockDim.x + threadIdx.x;
  if (e >= Etot) return;
  int s, d; edge_pair(src, dst, E, e, s, d);
  for (int hh = 0; hh < H; ++hh) {
    float v = es[(size_t)s * 8 + hh] + ed[(size_t)d * 8 + hh];
    v = v > 0.f ? v : 0.2f * v;
    atomicAdd(denom + (size_t)d * 8 + hh, __expf(v - m[(size_t)d * 8 + hh]));
  }
}

// one block per edge: agg[dst,h,c] += alpha[h] * h[src,h,c]  (f32 L2-side atomics)
__global__ __launch_bounds__(256) void k_edge_agg(
    const int* __restrict__ src, const int* __restrict__ dst, int E,
    const float* __restrict__ es, const float* __restrict__ ed,
    const float* __restrict__ m, const float* __restrict__ denom,
    const float* __restrict__ h, int ldh,
    float* __restrict__ agg, int ldagg, int H, int C, int D) {
  __shared__ float alpha[8];
  __shared__ int sSrc, sDst;
  int e = blockIdx.x;
  if (threadIdx.x == 0) { int s, d; edge_pair(src, dst, E, e, s, d); sSrc = s; sDst = d; }
  __syncthreads();
  int s = sSrc, d = sDst;
  if ((int)threadIdx.x < H) {
    int hh = threadIdx.x;
    float v = es[(size_t)s * 8 + hh] + ed[(size_t)d * 8 + hh];
    v = v > 0.f ? v : 0.2f * v;
    alpha[hh] = __expf(v - m[(size_t)d * 8 + hh]) / fmaxf(denom[(size_t)d * 8 + hh], 1e-16f);
  }
  __syncthreads();
  const float* hs = h + (size_t)s * ldh;
  float* ad = agg + (size_t)d * ldagg;
  if ((D & 3) == 0) {           // vectorized path: b128 gather + 4 atomics
    int D4 = D >> 2;
    const float4* hs4 = (const float4*)hs;
    for (int c4 = threadIdx.x; c4 < D4; c4 += blockDim.x) {
      float4 v4 = hs4[c4];
      int c = c4 << 2;
      float a0 = alpha[c / C];
      float a3 = alpha[(c + 3) / C];   // C multiple of 4 in layers 1-2 => a0==a3
      atomicAdd(ad + c + 0, a0 * v4.x);
      atomicAdd(ad + c + 1, a0 * v4.y);
      atomicAdd(ad + c + 2, a0 * v4.z);
      atomicAdd(ad + c + 3, a3 * v4.w);
    }
  } else {
    for (int c = threadIdx.x; c < D; c += blockDim.x)
      atomicAdd(ad + c, alpha[c / C] * hs[c]);
  }
}

// ---------------- epilogues ----------------
// x_next = elu(agg + skip + b + bl) -> bf16 activation for next layer (stride D)
__global__ __launch_bounds__(256) void k_epilogue12(
    const float* __restrict__ agg, const float* __restrict__ skip,
    const float* __restrict__ b, const float* __restrict__ bl,
    unsigned short* __restrict__ xb, int N, int D) {
  size_t i = (size_t)blockIdx.x * blockDim.x + threadIdx.x;
  if (i >= (size_t)N * D) return;
  int dcol = (int)(i % D);
  float v = agg[i] + skip[i] + b[dcol] + bl[dcol];
  v = v > 0.f ? v : __expf(v) - 1.f;            // elu
  xb[i] = f2bf(v);
}

// out = mean_h(agg) + b3 + skip + bl3
__global__ __launch_bounds__(256) void k_epilogue3(
    const float* __restrict__ agg, int ldagg,
    const float* __restrict__ skip, int ldskip,
    const float* __restrict__ b3, const float* __restrict__ bl3,
    float* __restrict__ out, int N, int C, int H) {
  int i = blockIdx.x * blockDim.x + threadIdx.x;
  if (i >= N * C) return;
  int n = i / C, c = i - n * C;
  float s = 0.f;
  for (int hh = 0; hh < H; ++hh) s += agg[(size_t)n * ldagg + hh * C + c];
  out[i] = s / (float)H + b3[c] + skip[(size_t)n * ldskip + c] + bl3[c];
}

// ---------------- host side ----------------
static inline unsigned nblk(size_t n, int t) { return (unsigned)((n + t - 1) / t); }

static void run_attention(hipStream_t stream, const int* srcI, const int* dstI, int E, int Etot,
                          const float* hbuf, int ldh, const float* asrc, const float* adst,
                          float* es, float* ed, float* mx, float* dn, float* agg, int ldagg,
                          int N, int H, int C, int D) {
  const int TPB = 256;
  k_attn_coef<<<nblk((size_t)N * H, 8), TPB, 0, stream>>>(hbuf, ldh, asrc, adst, es, ed, N, H, C);
  k_fill<<<nblk((size_t)N * 8, TPB), TPB, 0, stream>>>(mx, -3.0e38f, (size_t)N * 8);
  k_fill<<<nblk((size_t)N * 8, TPB), TPB, 0, stream>>>(dn, 0.f, (size_t)N * 8);
  k_fill<<<nblk((size_t)N * ldagg, TPB), TPB, 0, stream>>>(agg, 0.f, (size_t)N * ldagg);
  unsigned eb = nblk((size_t)Etot, TPB);
  k_edge_max<<<eb, TPB, 0, stream>>>(srcI, dstI, E, Etot, es, ed, mx, H);
  k_edge_sum<<<eb, TPB, 0, stream>>>(srcI, dstI, E, Etot, es, ed, mx, dn, H);
  k_edge_agg<<<(unsigned)Etot, TPB, 0, stream>>>(srcI, dstI, E, es, ed, mx, dn,
                                                 hbuf, ldh, agg, ldagg, H, C, D);
}

extern "C" void kernel_launch(void* const* d_in, const int* in_sizes, int n_in,
                              void* d_out, int out_size, void* d_ws, size_t ws_size,
                              hipStream_t stream) {
  (void)n_in; (void)out_size; (void)ws_size;
  const int F_IN = 50, D1 = 1024, H1 = 4, C1 = 256, C3 = 121, H3 = 6;
  const int D3 = H3 * C3;   // 726
  const int D3p = 736;      // 23 * 32 (padded)
  const int Cl3p = 128;     // skip3 padded cols (4 * 32)
  const int N = in_sizes[0] / F_IN;
  const int E = in_sizes[1] / 2;
  const int Etot = E + N;

  const float* x    = (const float*)d_in[0];
  const int*   ei   = (const int*)d_in[1];
  const int*   srcI = ei;
  const int*   dstI = ei + E;
  const float* W1  = (const float*)d_in[2];
  const float* a1s = (const float*)d_in[3];
  const float* a1d = (const float*)d_in[4];
  const float* b1  = (const float*)d_in[5];
  const float* Wl1 = (const float*)d_in[6];
  const float* bl1 = (const float*)d_in[7];
  const float* W2  = (const float*)d_in[8];
  const float* a2s = (const float*)d_in[9];
  const float* a2d = (const float*)d_in[10];
  const float* b2  = (const float*)d_in[11];
  const float* Wl2 = (const float*)d_in[12];
  const float* bl2 = (const float*)d_in[13];
  const float* W3  = (const float*)d_in[14];
  const float* a3s = (const float*)d_in[15];
  const float* a3d = (const float*)d_in[16];
  const float* b3  = (const float*)d_in[17];
  const float* Wl3 = (const float*)d_in[18];
  const float* bl3 = (const float*)d_in[19];
  float* out = (float*)d_out;

  size_t cur = 0;
  auto take = [&](size_t bytes) -> char* {
    cur = (cur + 255) & ~(size_t)255;
    char* p = (char*)d_ws + cur;
    cur += bytes;
    return p;
  };
  unsigned short* xb  = (unsigned short*)take((size_t)N * D1 * 2);   // bf16 activations
  unsigned short* Wb  = (unsigned short*)take((size_t)1024 * 1024 * 2);
  unsigned short* Wlb = (unsigned short*)take((size_t)1024 * 1024 * 2);
  float* hbuf = (float*)take((size_t)N * D1 * 4);
  float* skip = (float*)take((size_t)N * D1 * 4);
  float* agg  = (float*)take((size_t)N * D1 * 4);
  float* es   = (float*)take((size_t)N * 8 * 4);
  float* ed   = (float*)take((size_t)N * 8 * 4);
  float* mx   = (float*)take((size_t)N * 8 * 4);
  float* dn   = (float*)take((size_t)N * 8 * 4);

  const int TPB = 256;
  const int Mt2 = (N + 31) / 32;   // 625 (exact: N = 20000)

  // ---------------- Layer 1 (K=50 -> pad 64) ----------------
  k_convert_x<<<nblk((size_t)N * 64, TPB), TPB, 0, stream>>>(x, xb, N, F_IN, 64);
  k_convert_w<<<nblk((size_t)D1 * 64, TPB), TPB, 0, stream>>>(W1, Wb, F_IN, D1, 64, D1);
  k_convert_w<<<nblk((size_t)D1 * 64, TPB), TPB, 0, stream>>>(Wl1, Wlb, F_IN, D1, 64, D1);
  {
    int Nt2 = D1 / 32;
    unsigned g = nblk((size_t)Mt2 * Nt2, 8);
    k_gemm_wmma<<<g, TPB, 0, stream>>>(xb, 64, Wb, 64, hbuf, D1, Mt2, Nt2, 64);
    k_gemm_wmma<<<g, TPB, 0, stream>>>(xb, 64, Wlb, 64, skip, D1, Mt2, Nt2, 64);
  }
  run_attention(stream, srcI, dstI, E, Etot, hbuf, D1, a1s, a1d, es, ed, mx, dn,
                agg, D1, N, H1, C1, D1);
  k_epilogue12<<<nblk((size_t)N * D1, TPB), TPB, 0, stream>>>(agg, skip, b1, bl1, xb, N, D1);

  // ---------------- Layer 2 (K=1024) ----------------
  k_convert_w<<<nblk((size_t)D1 * D1, TPB), TPB, 0, stream>>>(W2, Wb, D1, D1, D1, D1);
  k_convert_w<<<nblk((size_t)D1 * D1, TPB), TPB, 0, stream>>>(Wl2, Wlb, D1, D1, D1, D1);
  {
    int Nt2 = D1 / 32;
    unsigned g = nblk((size_t)Mt2 * Nt2, 8);
    k_gemm_wmma<<<g, TPB, 0, stream>>>(xb, D1, Wb, D1, hbuf, D1, Mt2, Nt2, D1);
    k_gemm_wmma<<<g, TPB, 0, stream>>>(xb, D1, Wlb, D1, skip, D1, Mt2, Nt2, D1);
  }
  run_attention(stream, srcI, dstI, E, Etot, hbuf, D1, a2s, a2d, es, ed, mx, dn,
                agg, D1, N, H1, C1, D1);
  k_epilogue12<<<nblk((size_t)N * D1, TPB), TPB, 0, stream>>>(agg, skip, b2, bl2, xb, N, D1);

  // ---------------- Layer 3 (K=1024, out 726 -> pad 736; skip 121 -> pad 128) ----------------
  k_convert_w<<<nblk((size_t)D3p * D1, TPB), TPB, 0, stream>>>(W3, Wb, D1, D3, D1, D3p);
  k_convert_w<<<nblk((size_t)Cl3p * D1, TPB), TPB, 0, stream>>>(Wl3, Wlb, D1, C3, D1, Cl3p);
  {
    int Nt2 = D3p / 32;
    unsigned g = nblk((size_t)Mt2 * Nt2, 8);
    k_gemm_wmma<<<g, TPB, 0, stream>>>(xb, D1, Wb, D1, hbuf, D3p, Mt2, Nt2, D1);
  }
  {
    int Nt2 = Cl3p / 32;
    unsigned g = nblk((size_t)Mt2 * Nt2, 8);
    k_gemm_wmma<<<g, TPB, 0, stream>>>(xb, D1, Wlb, D1, skip, Cl3p, Mt2, Nt2, D1);
  }
  run_attention(stream, srcI, dstI, E, Etot, hbuf, D3p, a3s, a3d, es, ed, mx, dn,
                agg, D3p, N, H3, C3, D3);
  k_epilogue3<<<nblk((size_t)N * C3, TPB), TPB, 0, stream>>>(agg, D3p, skip, Cl3p,
                                                            b3, bl3, out, N, C3, H3);
}
